// MoELayer_30090540876253
// MI455X (gfx1250) — matrix operand order repaired and checked by reading, not compiled, not run on hardware
//
#include <hip/hip_runtime.h>
#include <hip/hip_bf16.h>

#define DDIM 1024
#define HDIM 4096
#define NEXP 8
#define NTOK 4096
#define NROW (NTOK * 2)          // top-2: exactly 2 rows per token
#define BM 128
#define BN 128
#define KB 32
#define LDSTRIDE (KB + 8)        // 40 bf16 = 80B row stride (16B aligned, bank-spread)
#define MAXRB (NROW / BM + NEXP) // 64 + 8 = 72 worst-case row blocks

typedef __bf16 bf16_t;
typedef __attribute__((ext_vector_type(8)))  bf16_t v8bf;
typedef __attribute__((ext_vector_type(16))) bf16_t v16bf;
typedef __attribute__((ext_vector_type(8)))  float  v8f;
typedef __attribute__((ext_vector_type(4)))  float  v4f;

// ---------------------------------------------------------------- router
__global__ void moe_router(const float* __restrict__ x, const float* __restrict__ Wg,
                           const float* __restrict__ bg,
                           int* __restrict__ topkIdx, float* __restrict__ topkW) {
  const int wave = threadIdx.x >> 5;
  const int lane = threadIdx.x & 31;
  const int t = blockIdx.x * 8 + wave;
  const float* xt = x + (size_t)t * DDIM;

  float acc[NEXP];
#pragma unroll
  for (int e = 0; e < NEXP; ++e) acc[e] = 0.0f;
  for (int d = lane; d < DDIM; d += 32) {
    float xv = xt[d];
#pragma unroll
    for (int e = 0; e < NEXP; ++e) acc[e] += xv * Wg[e * DDIM + d];
  }
#pragma unroll
  for (int e = 0; e < NEXP; ++e) {
#pragma unroll
    for (int off = 16; off >= 1; off >>= 1) acc[e] += __shfl_xor(acc[e], off, 32);
  }
  if (lane == 0) {
    float lg[NEXP], m = -3.0e38f;
#pragma unroll
    for (int e = 0; e < NEXP; ++e) { lg[e] = acc[e] + bg[e]; m = fmaxf(m, lg[e]); }
    float w[NEXP], s = 0.0f;
#pragma unroll
    for (int e = 0; e < NEXP; ++e) { w[e] = expf(lg[e] - m); s += w[e]; }
    float inv = 1.0f / s;
#pragma unroll
    for (int e = 0; e < NEXP; ++e) w[e] *= inv;
    int i0 = 0;
#pragma unroll
    for (int e = 1; e < NEXP; ++e) if (w[e] > w[i0]) i0 = e;
    int i1 = (i0 == 0) ? 1 : 0;
#pragma unroll
    for (int e = 0; e < NEXP; ++e) if (e != i0 && w[e] > w[i1]) i1 = e;
    topkIdx[2 * t] = i0;     topkIdx[2 * t + 1] = i1;
    topkW[2 * t]  = w[i0];   topkW[2 * t + 1]  = w[i1];
  }
}

// --------------------------------------------------- routing: count/scan/assign
__global__ void moe_route_count(const int* __restrict__ topkIdx, int* __restrict__ counts) {
  const int e = blockIdx.x, lane = threadIdx.x;
  int c = 0;
  for (int i = lane; i < NROW; i += 32) c += (topkIdx[i] == e) ? 1 : 0;
#pragma unroll
  for (int off = 16; off >= 1; off >>= 1) c += __shfl_xor(c, off, 32);
  if (lane == 0) counts[e] = c;
}

__global__ void moe_route_scan(const int* __restrict__ counts,
                               int* __restrict__ offsets, int* __restrict__ blockStart) {
  if (threadIdx.x == 0) {
    int o = 0, b = 0;
    for (int e = 0; e < NEXP; ++e) {
      offsets[e] = o; blockStart[e] = b;
      o += counts[e]; b += (counts[e] + BM - 1) / BM;
    }
    offsets[NEXP] = o; blockStart[NEXP] = b;
  }
}

__global__ void moe_route_assign(const int* __restrict__ topkIdx, const int* __restrict__ offsets,
                                 int* __restrict__ tokenOf, int* __restrict__ rowOf) {
  const int e = blockIdx.x, lane = threadIdx.x;
  int base = offsets[e];
  for (int i0 = 0; i0 < NROW; i0 += 32) {
    const int i = i0 + lane;
    const bool sel = (topkIdx[i] == e);
    const unsigned mask = __builtin_amdgcn_ballot_w32(sel);
    if (sel) {
      const int pos = base + __popc(mask & ((1u << lane) - 1u));
      tokenOf[pos] = i >> 1;
      rowOf[i] = pos;
    }
    base += __popc(mask);
  }
}

// ---------------------------------------------------------------- helpers
__device__ __forceinline__ void cvtStore16(bf16_t* dst, const v4f s[4]) {
  v8bf lo, h8;
#pragma unroll
  for (int j = 0; j < 4; ++j) {
    lo[j]     = (bf16_t)s[0][j];
    lo[j + 4] = (bf16_t)s[1][j];
    h8[j]     = (bf16_t)s[2][j];
    h8[j + 4] = (bf16_t)s[3][j];
  }
  *(v8bf*)dst       = lo;
  *(v8bf*)(dst + 8) = h8;
}

// 32-byte global -> LDS async copy (ASYNCcnt-tracked, CDNA5).
// IOFFSET is added to BOTH the global and LDS addresses (ISA 15.18.3), so one
// base pair covers both 16B beats.
__device__ __forceinline__ void asyncCopy32B(void* ldsDst, const void* gsrc) {
  // generic LDS pointer = {shared aperture, lds_offset}; low 32 bits are the LDS address
  unsigned l = (unsigned)(unsigned long long)ldsDst;
  asm volatile("global_load_async_to_lds_b128 %0, %1, off\n\t"
               "global_load_async_to_lds_b128 %0, %1, off offset:16"
               :: "v"(l), "v"(gsrc) : "memory");
}
__device__ __forceinline__ void asyncWait0() {
  asm volatile("s_wait_asynccnt 0x0" ::: "memory");
}

// ---------------------------------------------------------------- grouped GEMM
template <int KD, int ND, bool A_BF16, bool DO_GELU, bool OUT_BF16>
__global__ __launch_bounds__(256) void moe_ffn_gemm(
    const float* __restrict__ Afp,   // [NTOK, KD] (gathered via tokenOf) when !A_BF16
    const bf16_t* __restrict__ Abf,  // [NROW, KD] when A_BF16
    const float* __restrict__ W,     // [E, ND, KD]
    const float* __restrict__ bias,  // [E, ND]
    const int* __restrict__ tokenOf,
    const int* __restrict__ offsets,
    const int* __restrict__ blockStart,
    bf16_t* __restrict__ outBf,      // [NROW, ND]
    float* __restrict__ outFp)       // [NROW, ND]
{
  __shared__ bf16_t As[2][BM][LDSTRIDE];
  __shared__ bf16_t Bs[2][BM][LDSTRIDE];

  const int rb = blockIdx.y;
  int expert = -1;
#pragma unroll
  for (int e = 0; e < NEXP; ++e)
    if (rb >= blockStart[e] && rb < blockStart[e + 1]) expert = e;
  if (expert < 0) return;  // uniform exit

  const int rowsStart = offsets[expert] + (rb - blockStart[expert]) * BM;
  const int mValid = min(BM, offsets[expert + 1] - rowsStart);

  const int tid  = threadIdx.x;
  const int lane = tid & 31;
  const int waveId = tid >> 5;
  const int wm0 = (waveId >> 1) * 32;  // wave's 32-row slice of the 128-row tile
  const int wn0 = (waveId & 1) * 64;   // wave's 64-col slice of the 128-col tile
  const int lr = lane & 15;
  const int hi = lane >> 4;
  const int aC0 = hi * 8;              // A frag K chunks: {aC0, aC0+16}  (ISA 16-bit A layout)
  const int bC0 = hi * 8;              // B frag K chunks: {bC0, bC0+16}  (ISA 16-bit B layout)

  const int n0 = blockIdx.x * BN;

  // tile loaders: 2 threads per row, 16 K-elements each.
  // Invalid rows are clamped to a valid row: their results only land in output
  // rows >= mValid, which the epilogue masks out (no zero-fill needed).
  const int ldRow = tid >> 1;
  const int ldK = (tid & 1) * 16;
  const int srcRow = rowsStart + min(ldRow, mValid - 1);
  const bf16_t* aB = A_BF16 ? (Abf + (size_t)srcRow * KD + ldK) : nullptr;
  const float*  aF = A_BF16 ? nullptr : (Afp + (size_t)tokenOf[srcRow] * KD + ldK);
  const float*  wRow = W + ((size_t)expert * ND + (n0 + ldRow)) * KD + ldK;

  v8f acc[2][4];
#pragma unroll
  for (int mi = 0; mi < 2; ++mi)
#pragma unroll
    for (int ni = 0; ni < 4; ++ni)
#pragma unroll
      for (int i = 0; i < 8; ++i) acc[mi][ni][i] = 0.0f;

  // ---- prologue: stage tile k=0 ----
  v4f aStage[4], bStage[4];
#pragma unroll
  for (int j = 0; j < 4; ++j) bStage[j] = *(const v4f*)(wRow + 4 * j);
  if constexpr (A_BF16) {
    asyncCopy32B(&As[0][ldRow][ldK], aB);           // pure bf16 copy -> async to LDS
  } else {
#pragma unroll
    for (int j = 0; j < 4; ++j) aStage[j] = *(const v4f*)(aF + 4 * j);
  }

  int buf = 0;
  for (int k0 = 0; k0 < KD; k0 += KB) {
    // ---- commit tile k into LDS[buf] ----
    cvtStore16(&Bs[buf][ldRow][ldK], bStage);
    if constexpr (A_BF16) {
      asyncWait0();                                  // async fill of As[buf] done
    } else {
      cvtStore16(&As[buf][ldRow][ldK], aStage);
    }
    __syncthreads();

    // ---- start tile k+1 into the back buffer / stage regs ----
    if (k0 + KB < KD) {
      if constexpr (A_BF16) {
        asyncCopy32B(&As[buf ^ 1][ldRow][ldK], aB + k0 + KB);
      } else {
#pragma unroll
        for (int j = 0; j < 4; ++j) aStage[j] = *(const v4f*)(aF + k0 + KB + 4 * j);
      }
#pragma unroll
      for (int j = 0; j < 4; ++j) bStage[j] = *(const v4f*)(wRow + k0 + KB + 4 * j);
      if (k0 + 2 * KB < KD) __builtin_prefetch(wRow + k0 + 2 * KB, 0, 0);
    }

    // ---- fragments + WMMA on LDS[buf] ----
    v16bf aFr[2], bFr[4];
#pragma unroll
    for (int mi = 0; mi < 2; ++mi) {
      const int r = wm0 + mi * 16 + lr;
      v8bf x0 = *(const v8bf*)&As[buf][r][aC0];
      v8bf x1 = *(const v8bf*)&As[buf][r][aC0 + 16];
      aFr[mi] = __builtin_shufflevector(x0, x1, 0,1,2,3,4,5,6,7,8,9,10,11,12,13,14,15);
    }
#pragma unroll
    for (int ni = 0; ni < 4; ++ni) {
      const int r = wn0 + ni * 16 + lr;
      v8bf x0 = *(const v8bf*)&Bs[buf][r][bC0];
      v8bf x1 = *(const v8bf*)&Bs[buf][r][bC0 + 16];
      bFr[ni] = __builtin_shufflevector(x0, x1, 0,1,2,3,4,5,6,7,8,9,10,11,12,13,14,15);
    }
#pragma unroll
    for (int mi = 0; mi < 2; ++mi)
#pragma unroll
      for (int ni = 0; ni < 4; ++ni)
        acc[mi][ni] = __builtin_amdgcn_wmma_f32_16x16x32_bf16(
            false, aFr[mi], false, bFr[ni], (short)0, acc[mi][ni], false, false);
    buf ^= 1;
  }

  // ---- epilogue: bias (+GELU) and store ----
#pragma unroll
  for (int mi = 0; mi < 2; ++mi) {
#pragma unroll
    for (int ni = 0; ni < 4; ++ni) {
      const int col = n0 + wn0 + ni * 16 + lr;
      const float bv = bias[(size_t)expert * ND + col];
#pragma unroll
      for (int i = 0; i < 8; ++i) {
        const int rLoc = wm0 + mi * 16 + (hi ? (8 + i) : i);
        if (rLoc < mValid) {
          float v = acc[mi][ni][i] + bv;
          if (DO_GELU) v = 0.5f * v * (1.0f + erff(v * 0.70710678118654752f));
          const size_t idx = (size_t)(rowsStart + rLoc) * ND + col;
          if (OUT_BF16) outBf[idx] = (bf16_t)v; else outFp[idx] = v;
        }
      }
    }
  }
}

// ---------------------------------------------------------------- combine
__global__ void moe_combine(const float* __restrict__ y, const float* __restrict__ topkW,
                            const int* __restrict__ rowOf, float* __restrict__ out) {
  const int t = blockIdx.x;
  const int o = threadIdx.x * 4;
  const int r0 = rowOf[2 * t], r1 = rowOf[2 * t + 1];
  const float w0 = topkW[2 * t], w1 = topkW[2 * t + 1];
  v4f y0 = *(const v4f*)(y + (size_t)r0 * DDIM + o);
  v4f y1 = *(const v4f*)(y + (size_t)r1 * DDIM + o);
  v4f r = w0 * y0 + w1 * y1;
  *(v4f*)(out + (size_t)t * DDIM + o) = r;
}

// ---------------------------------------------------------------- launch
extern "C" void kernel_launch(void* const* d_in, const int* in_sizes, int n_in,
                              void* d_out, int out_size, void* d_ws, size_t ws_size,
                              hipStream_t stream) {
  const float* x  = (const float*)d_in[0];
  const float* Wg = (const float*)d_in[1];
  const float* bg = (const float*)d_in[2];
  const float* W1 = (const float*)d_in[3];
  const float* b1 = (const float*)d_in[4];
  const float* W2 = (const float*)d_in[5];
  const float* b2 = (const float*)d_in[6];
  float* out = (float*)d_out;
  (void)in_sizes; (void)n_in; (void)out_size; (void)ws_size;

  char* ws = (char*)d_ws;
  size_t off = 0;
  auto carve = [&](size_t bytes) -> char* {
    char* p = ws + off;
    off = (off + bytes + 255) & ~(size_t)255;
    return p;
  };
  int*    topkIdx    = (int*)carve(NROW * sizeof(int));
  float*  topkW      = (float*)carve(NROW * sizeof(float));
  int*    counts     = (int*)carve(NEXP * sizeof(int));
  int*    offsets    = (int*)carve((NEXP + 1) * sizeof(int));
  int*    blockStart = (int*)carve((NEXP + 1) * sizeof(int));
  int*    tokenOf    = (int*)carve(NROW * sizeof(int));
  int*    rowOf      = (int*)carve(NROW * sizeof(int));
  bf16_t* hbuf       = (bf16_t*)carve((size_t)NROW * HDIM * sizeof(bf16_t));
  float*  ybuf       = (float*)carve((size_t)NROW * DDIM * sizeof(float));

  moe_router<<<NTOK / 8, 256, 0, stream>>>(x, Wg, bg, topkIdx, topkW);
  moe_route_count<<<NEXP, 32, 0, stream>>>(topkIdx, counts);
  moe_route_scan<<<1, 32, 0, stream>>>(counts, offsets, blockStart);
  moe_route_assign<<<NEXP, 32, 0, stream>>>(topkIdx, offsets, tokenOf, rowOf);

  moe_ffn_gemm<DDIM, HDIM, false, true, true>
      <<<dim3(HDIM / BN, MAXRB), 256, 0, stream>>>(
          x, nullptr, W1, b1, tokenOf, offsets, blockStart, hbuf, nullptr);
  moe_ffn_gemm<HDIM, DDIM, true, false, false>
      <<<dim3(DDIM / BN, MAXRB), 256, 0, stream>>>(
          nullptr, hbuf, W2, b2, tokenOf, offsets, blockStart, nullptr, ybuf);

  moe_combine<<<NTOK, 256, 0, stream>>>(ybuf, topkW, rowOf, out);
}
